// SelfAttention_68831145886674
// MI455X (gfx1250) — matrix operand order repaired and checked by reading, not compiled, not run on hardware
//
#include <hip/hip_runtime.h>

#define DEVI __device__ __forceinline__

typedef __attribute__((ext_vector_type(16))) __bf16 v16bf;
typedef __attribute__((ext_vector_type(8)))  float  v8f;
typedef __attribute__((ext_vector_type(4)))  unsigned int u32x4;
typedef __attribute__((ext_vector_type(8)))  int i32x8;
typedef __attribute__((ext_vector_type(4)))  int i32x4;

static constexpr int NPIX = 256 * 256;

DEVI v8f wmma_bf16(v16bf a, v16bf b, v8f c) {
  return __builtin_amdgcn_wmma_f32_16x16x32_bf16(false, a, false, b, (short)0, c, false, false);
}

// A operand: 16x32 bf16, lane l holds row (l&15); lanes 16-31 carry K {8..15,24..31}
template <typename F>
DEVI v16bf make_a_frag(int lane, F f) {
  const int row = lane & 15;
  const int hi8 = (lane & 16) ? 8 : 0;
  v16bf a;
#pragma unroll
  for (int p = 0; p < 8; ++p) {
    const int k = ((p >= 4) ? 16 : 0) + hi8 + 2 * (p & 3);
    a[2 * p]     = f(row, k);
    a[2 * p + 1] = f(row, k + 1);
  }
  return a;
}

// B operand: 32x16 bf16, lane l holds column (l&15); lanes 16-31 carry K {8..15,24..31}
template <typename F>
DEVI v16bf make_b_frag(int lane, F f) {
  const int col = lane & 15;
  const int kh  = (lane & 16) ? 8 : 0;
  v16bf b;
#pragma unroll
  for (int v = 0; v < 8; ++v) {
    const int k = ((v >= 4) ? 16 : 0) + kh + 2 * (v & 3);
    b[2 * v]     = f(k, col);
    b[2 * v + 1] = f(k + 1, col);
  }
  return b;
}

// ---------------------------------------------------------------------------
// Phase 0: zero reduction workspace
// ---------------------------------------------------------------------------
__global__ void zero_ws(float* p, int n) {
  int i = blockIdx.x * 256 + threadIdx.x;
  if (i < n) p[i] = 0.f;
}

// ---------------------------------------------------------------------------
// Phase 1: per (batch, branch) reduction of Ksum, Vsum, matrix = K * V^T
// x tiles are brought into LDS by the Tensor Data Mover (TDM), which also
// applies the +4-dword row padding (pad_interval=32dw, pad_amount=4dw) that
// gives the T1LD=36 LDS pitch.
// ---------------------------------------------------------------------------
static constexpr int T1 = 32;          // pixels per tile
static constexpr int T1LD = T1 + 4;    // = TDM row pitch in dwords
static constexpr int WG_PER_TGT = 32;  // workgroups per (b, branch)

__global__ __launch_bounds__(256) void p1_kv_reduce(
    const float* __restrict__ x1, const float* __restrict__ x2,
    const float* __restrict__ k1w, const float* __restrict__ k1b,
    const float* __restrict__ v1w, const float* __restrict__ v1b,
    const float* __restrict__ k2w, const float* __restrict__ k2b,
    const float* __restrict__ v2w, const float* __restrict__ v2b,
    float* __restrict__ gK, float* __restrict__ gV, float* __restrict__ gM)
{
  __shared__ float  Xs[64 * T1LD];
  __shared__ float  Ksf[64 * T1LD];
  __shared__ __bf16 Kb[64 * T1LD];
  __shared__ __bf16 Vb[64 * T1LD];
  __shared__ float  sK[64], sV[64];

  const int tgt = blockIdx.x / WG_PER_TGT;       // 0..15 = (batch, branch)
  const int wg  = blockIdx.x % WG_PER_TGT;
  const int bb  = tgt >> 1, br = tgt & 1;
  const float* x     = (br ? x2 : x1) + (size_t)bb * 64 * NPIX;
  const float* kw    = br ? k2w : k1w;
  const float* kbias = br ? k2b : k1b;
  const float* vw    = br ? v2w : v1w;
  const float* vbias = br ? v2b : v1b;

  const int tid = threadIdx.x, lane = tid & 31, wid = tid >> 5;
  const int hi = lane >> 4, col = lane & 15;

  if (tid < 64) { sK[tid] = 0.f; sV[tid] = 0.f; }

  // Waves 0-3 compute K m-blocks, waves 4-7 compute V m-blocks.
  const int proj = wid >> 2, mb = wid & 3;
  const float* W  = proj ? vw : kw;
  const float* Bv = proj ? vbias : kbias;
  v16bf aW[2];
#pragma unroll
  for (int ks = 0; ks < 2; ++ks)
    aW[ks] = make_a_frag(lane, [&](int r, int k) -> __bf16 {
      return (__bf16)W[(mb * 16 + r) * 64 + ks * 32 + k]; });
  float biasv[8];
#pragma unroll
  for (int i = 0; i < 8; ++i) biasv[i] = Bv[mb * 16 + i + 8 * hi];

  // TDM descriptor group 1 (constant across tiles):
  //  data_size=2 (4B), pad_enable=1, pad_interval=4 (32 dwords), pad_amount=3
  //  (4 dwords), tensor_dim0=NPIX, tensor_dim1=64, tile=32x64, stride0=NPIX.
  i32x8 g1c;
  g1c[0] = (int)((2u << 16) | (1u << 20) | (4u << 22) | (3u << 25));
  g1c[1] = 0;                                    // atomic addr 0, dim0 lo16 = 0
  g1c[2] = (int)((64u << 16) | 1u);              // dim0 hi16 = 1, dim1 lo16 = 64
  g1c[3] = (int)(32u << 16);                     // dim1 hi16 = 0, tile_dim0 = 32
  g1c[4] = (int)64;                              // tile_dim1 = 64, tile_dim2 = 0
  g1c[5] = (int)NPIX;                            // dim0_stride lo32
  g1c[6] = 0;                                    // dim0_stride hi16, dim1_stride lo16
  g1c[7] = 0;
  const i32x4 gz = {0, 0, 0, 0};
  const i32x8 gz8 = {0, 0, 0, 0, 0, 0, 0, 0};
  const unsigned ldsXs = (unsigned)(uintptr_t)(&Xs[0]);   // LDS aperture offset
  const unsigned long long xbase = (unsigned long long)(uintptr_t)x;

  // Each wave owns two 16x16 fragments of the 64x64 matrix accumulator.
  const int f0 = wid * 2, f1 = wid * 2 + 1;
  v8f m0 = {}, m1 = {};

  for (int tile = wg; tile < NPIX / T1; tile += WG_PER_TGT) {
    const int n0 = tile * T1;
    __syncthreads();                 // everyone done reading Xs/Kb/Vb
    if (wid == 0) {                  // single wave issues the tensor DMA
      const unsigned long long ga = xbase + (unsigned long long)n0 * 4ull;
      u32x4 g0;
      g0[0] = 1u;                                          // count=1, user D#
      g0[1] = ldsXs;                                       // lds_addr
      g0[2] = (unsigned)ga;                                // global_addr lo32
      g0[3] = (unsigned)((ga >> 32) & 0x1ffffffu) | (2u << 30);  // hi25 | type=2
      __builtin_amdgcn_tensor_load_to_lds(g0, g1c, gz, gz, gz8, 0);
      __builtin_amdgcn_s_wait_tensorcnt(0);
    }
    __syncthreads();                 // publish Xs tile to all waves

    // projection: 16x32 result tile per wave (two n-blocks)
    v8f acc0 = {}, acc1 = {};
#pragma unroll
    for (int ks = 0; ks < 2; ++ks) {
      v16bf bm0 = make_b_frag(lane, [&](int k, int c) -> __bf16 {
        return (__bf16)Xs[(ks * 32 + k) * T1LD + c]; });
      v16bf bm1 = make_b_frag(lane, [&](int k, int c) -> __bf16 {
        return (__bf16)Xs[(ks * 32 + k) * T1LD + 16 + c]; });
      acc0 = wmma_bf16(aW[ks], bm0, acc0);
      acc1 = wmma_bf16(aW[ks], bm1, acc1);
    }
#pragma unroll
    for (int i = 0; i < 8; ++i) {
      const int row = mb * 16 + i + 8 * hi;
      const float v0 = acc0[i] + biasv[i];
      const float v1 = acc1[i] + biasv[i];
      if (proj) {
        Vb[row * T1LD + col]      = (__bf16)v0;
        Vb[row * T1LD + 16 + col] = (__bf16)v1;
      } else {
        Ksf[row * T1LD + col]      = v0;
        Ksf[row * T1LD + 16 + col] = v1;
      }
    }
    __syncthreads();

    // L2-normalize K per pixel (column), write bf16
    if (tid < T1) {
      float ss = 0.f;
      for (int r = 0; r < 64; ++r) { const float q = Ksf[r * T1LD + tid]; ss += q * q; }
      const float rn = rsqrtf(ss);
      for (int r = 0; r < 64; ++r)
        Kb[r * T1LD + tid] = (__bf16)(Ksf[r * T1LD + tid] * rn);
    }
    __syncthreads();

    // Ksum / Vsum partials
    if (tid < 64) {
      float a0 = 0.f, a1 = 0.f;
      for (int n = 0; n < T1; ++n) {
        a0 += (float)Kb[tid * T1LD + n];
        a1 += (float)Vb[tid * T1LD + n];
      }
      sK[tid] += a0; sV[tid] += a1;
    }

    // matrix += K_tile * V_tile^T  (K-dim = 32 pixels), LDS provides transpose
    v16bf aK0 = make_a_frag(lane, [&](int r, int k) -> __bf16 {
      return Kb[((f0 >> 2) * 16 + r) * T1LD + k]; });
    v16bf bV0 = make_b_frag(lane, [&](int k, int c) -> __bf16 {
      return Vb[((f0 & 3) * 16 + c) * T1LD + k]; });
    m0 = wmma_bf16(aK0, bV0, m0);
    v16bf aK1 = make_a_frag(lane, [&](int r, int k) -> __bf16 {
      return Kb[((f1 >> 2) * 16 + r) * T1LD + k]; });
    v16bf bV1 = make_b_frag(lane, [&](int k, int c) -> __bf16 {
      return Vb[((f1 & 3) * 16 + c) * T1LD + k]; });
    m1 = wmma_bf16(aK1, bV1, m1);
  }

  if (tid < 64) {
    atomicAdd(&gK[tgt * 64 + tid], sK[tid]);
    atomicAdd(&gV[tgt * 64 + tid], sV[tid]);
  }
  float* M = gM + (size_t)tgt * 4096;
#pragma unroll
  for (int i = 0; i < 8; ++i) {
    atomicAdd(&M[((f0 >> 2) * 16 + i + 8 * hi) * 64 + (f0 & 3) * 16 + col], m0[i]);
    atomicAdd(&M[((f1 >> 2) * 16 + i + 8 * hi) * 64 + (f1 & 3) * 16 + col], m1[i]);
  }
}

// ---------------------------------------------------------------------------
// Phase 2: fused per-pixel branch math + 3x3 conv; cat lives only in LDS.
// Output tile 16x16, halo 18x18 (324 pixels, padded to 21 n-blocks).
// ---------------------------------------------------------------------------
static constexpr int HALO = 18, HP = 324, NBQ = 21, PLD = 344, WLD = 392;
static constexpr size_t SMEM_P2 = 178016;

__global__ __launch_bounds__(256) void p2_fused(
    const float* __restrict__ x1, const float* __restrict__ x2,
    const float* __restrict__ q1w, const float* __restrict__ q1b,
    const float* __restrict__ r1w, const float* __restrict__ r1b,
    const float* __restrict__ q2w, const float* __restrict__ q2b,
    const float* __restrict__ r2w, const float* __restrict__ r2b,
    const float* __restrict__ gK, const float* __restrict__ gV,
    const float* __restrict__ gM,
    const float* __restrict__ catw, const float* __restrict__ catb,
    float* __restrict__ out)
{
  extern __shared__ char smem[];
  __bf16* catS  = (__bf16*)smem;                 // [128][PLD]
  __bf16* xsS   = (__bf16*)(smem + 88064);       // [64][PLD] (also wv, also conv W)
  __bf16* qsS   = (__bf16*)(smem + 132096);      // [64][PLD]
  float* denomS = (float*)(smem + 176128);       // [PLD]
  float* kseS   = (float*)(smem + 177504);       // [64]
  float* vsS    = (float*)(smem + 177760);       // [64]

  const int tid = threadIdx.x, lane = tid & 31, wid = tid >> 5;
  const int hi = lane >> 4, col = lane & 15;
  const int bx = blockIdx.x, by = blockIdx.y, bb = blockIdx.z;
  const int gx0 = bx * 16 - 1, gy0 = by * 16 - 1;

  for (int br = 0; br < 2; ++br) {
    const float* xp = (br ? x2 : x1) + (size_t)bb * 64 * NPIX;
    const float* qw = br ? q2w : q1w;
    const float* qb = br ? q2b : q1b;
    const float* rw = br ? r2w : r1w;
    const float* rb = br ? r2b : r1b;
    const int tgt = bb * 2 + br;
    const float* Mg = gM + (size_t)tgt * 4096;

    __syncthreads();
    if (tid < 64) {
      kseS[tid] = gK[tgt * 64 + tid] + 1e-6f;   // Ksum + EPS
      vsS[tid]  = gV[tgt * 64 + tid];
    }
    // halo x-tile load (64 * 324 = 81 * 256, uniform)
    for (int idx = tid; idx < 64 * HP; idx += 256) {
      const int c = idx / HP, p = idx - c * HP;
      const int py = p / HALO, px = p - py * HALO;
      const int gy = gy0 + py, gx = gx0 + px;
      float v = 0.f;
      if (gy >= 0 && gy < 256 && gx >= 0 && gx < 256)
        v = xp[(size_t)c * NPIX + gy * 256 + gx];
      xsS[c * PLD + p] = (__bf16)v;
    }
    __syncthreads();

    // Q projection (64x64 x 64xHP)
    for (int nb = wid; nb < NBQ; nb += 8) {
      v16bf b0 = make_b_frag(lane, [&](int k, int c) -> __bf16 {
        return xsS[k * PLD + nb * 16 + c]; });
      v16bf b1 = make_b_frag(lane, [&](int k, int c) -> __bf16 {
        return xsS[(32 + k) * PLD + nb * 16 + c]; });
#pragma unroll
      for (int mb = 0; mb < 4; ++mb) {
        v8f acc = {};
        v16bf a0 = make_a_frag(lane, [&](int r, int k) -> __bf16 {
          return (__bf16)qw[(mb * 16 + r) * 64 + k]; });
        acc = wmma_bf16(a0, b0, acc);
        v16bf a1 = make_a_frag(lane, [&](int r, int k) -> __bf16 {
          return (__bf16)qw[(mb * 16 + r) * 64 + 32 + k]; });
        acc = wmma_bf16(a1, b1, acc);
#pragma unroll
        for (int i = 0; i < 8; ++i) {
          const int row = mb * 16 + i + 8 * hi;
          qsS[row * PLD + nb * 16 + col] = (__bf16)(acc[i] + qb[row]);
        }
      }
    }
    __syncthreads();

    // per-pixel L2 norm of Q, denom = 1/(N + Qn . (Ksum+eps))
    for (int p = tid; p < HP; p += 256) {
      float ss = 0.f;
      for (int r = 0; r < 64; ++r) {
        const float q = (float)qsS[r * PLD + p];
        ss += q * q;
      }
      const float rn = rsqrtf(ss);
      float dq = 0.f;
      for (int r = 0; r < 64; ++r) {
        const float qn = (float)qsS[r * PLD + p] * rn;
        qsS[r * PLD + p] = (__bf16)qn;
        dq += qn * kseS[r];
      }
      denomS[p] = 1.f / ((float)NPIX + dq);
    }
    __syncthreads();

    // numer = vsum + M^T * Qn ; wv = numer * denom   -> reuse xsS
    for (int nb = wid; nb < NBQ; nb += 8) {
      v16bf b0 = make_b_frag(lane, [&](int k, int c) -> __bf16 {
        return qsS[k * PLD + nb * 16 + c]; });
      v16bf b1 = make_b_frag(lane, [&](int k, int c) -> __bf16 {
        return qsS[(32 + k) * PLD + nb * 16 + c]; });
#pragma unroll
      for (int mb = 0; mb < 4; ++mb) {
        v8f acc = {};
        v16bf a0 = make_a_frag(lane, [&](int r, int k) -> __bf16 {
          return (__bf16)Mg[k * 64 + mb * 16 + r]; });      // M transposed
        acc = wmma_bf16(a0, b0, acc);
        v16bf a1 = make_a_frag(lane, [&](int r, int k) -> __bf16 {
          return (__bf16)Mg[(32 + k) * 64 + mb * 16 + r]; });
        acc = wmma_bf16(a1, b1, acc);
#pragma unroll
        for (int i = 0; i < 8; ++i) {
          const int row = mb * 16 + i + 8 * hi;
          const int p = nb * 16 + col;
          if (p < HP)
            xsS[row * PLD + p] = (__bf16)((acc[i] + vsS[row]) * denomS[p]);
        }
      }
    }
    __syncthreads();

    // r projection -> catS (zero outside the image so conv SAME-pad is exact)
    for (int nb = wid; nb < NBQ; nb += 8) {
      v16bf b0 = make_b_frag(lane, [&](int k, int c) -> __bf16 {
        return xsS[k * PLD + nb * 16 + c]; });
      v16bf b1 = make_b_frag(lane, [&](int k, int c) -> __bf16 {
        return xsS[(32 + k) * PLD + nb * 16 + c]; });
#pragma unroll
      for (int mb = 0; mb < 4; ++mb) {
        v8f acc = {};
        v16bf a0 = make_a_frag(lane, [&](int r, int k) -> __bf16 {
          return (__bf16)rw[(mb * 16 + r) * 64 + k]; });
        acc = wmma_bf16(a0, b0, acc);
        v16bf a1 = make_a_frag(lane, [&](int r, int k) -> __bf16 {
          return (__bf16)rw[(mb * 16 + r) * 64 + 32 + k]; });
        acc = wmma_bf16(a1, b1, acc);
#pragma unroll
        for (int i = 0; i < 8; ++i) {
          const int row = mb * 16 + i + 8 * hi;
          const int p = nb * 16 + col;
          if (p < HP) {
            const int py = p / HALO, px = p - py * HALO;
            const int gy = gy0 + py, gx = gx0 + px;
            const bool inr = (gy >= 0 && gy < 256 && gx >= 0 && gx < 256);
            catS[(br * 64 + row) * PLD + p] = (__bf16)(inr ? acc[i] + rb[row] : 0.f);
          }
        }
      }
    }
  }

  // 3x3 conv as implicit GEMM: out[64,256] = W[64,1152] x im2col[1152,256]
  // k ordering: k = tap*128 + ic, so each 32-wide k-step has a single tap.
  v8f cacc[2][4] = {};
  __bf16* Wl = xsS;  // 64*WLD bf16 = 50176 B, overlays xsS+qsS (both free now)
  for (int chunk = 0; chunk < 3; ++chunk) {
    __syncthreads();
    for (int idx = tid; idx < 64 * 384; idx += 256) {   // 96*256, uniform
      const int o = idx / 384, k = idx - o * 384;
      const int kg = chunk * 384 + k;
      const int ic = kg & 127, tap = kg >> 7;
      Wl[o * WLD + k] = (__bf16)catw[(size_t)o * 1152 + ic * 9 + tap];
    }
    __syncthreads();
    for (int ksl = 0; ksl < 12; ++ksl) {
      const int ksg = chunk * 12 + ksl;
      const int tap = ksg >> 2, ic0 = (ksg & 3) * 32;
      const int ty = tap / 3, tx = tap - ty * 3;
      v16bf bf0 = make_b_frag(lane, [&](int k, int c) -> __bf16 {
        const int p = wid * 16 + c;
        const int py = p >> 4, px = p & 15;
        return catS[(ic0 + k) * PLD + (py + ty) * HALO + (px + tx)]; });
      v16bf bf1 = make_b_frag(lane, [&](int k, int c) -> __bf16 {
        const int p = (wid + 8) * 16 + c;
        const int py = p >> 4, px = p & 15;
        return catS[(ic0 + k) * PLD + (py + ty) * HALO + (px + tx)]; });
#pragma unroll
      for (int mbq = 0; mbq < 4; ++mbq) {
        v16bf a = make_a_frag(lane, [&](int r, int k) -> __bf16 {
          return Wl[(mbq * 16 + r) * WLD + ksl * 32 + k]; });
        cacc[0][mbq] = wmma_bf16(a, bf0, cacc[0][mbq]);
        cacc[1][mbq] = wmma_bf16(a, bf1, cacc[1][mbq]);
      }
    }
  }
#pragma unroll
  for (int j = 0; j < 2; ++j)
#pragma unroll
    for (int mbq = 0; mbq < 4; ++mbq)
#pragma unroll
      for (int i = 0; i < 8; ++i) {
        const int o = mbq * 16 + i + 8 * hi;
        const int p = (wid + 8 * j) * 16 + col;
        const int py = p >> 4, px = p & 15;
        out[((size_t)(bb * 64 + o)) * NPIX + (size_t)(by * 16 + py) * 256 + (bx * 16 + px)]
            = cacc[j][mbq][i] + catb[o];
      }
}

// ---------------------------------------------------------------------------
extern "C" void kernel_launch(void* const* d_in, const int* in_sizes, int n_in,
                              void* d_out, int out_size, void* d_ws, size_t ws_size,
                              hipStream_t stream) {
  (void)in_sizes; (void)n_in; (void)out_size; (void)ws_size;
  const float* x1  = (const float*)d_in[0];
  const float* x2  = (const float*)d_in[1];
  const float* q1w = (const float*)d_in[2];
  const float* q1b = (const float*)d_in[3];
  const float* k1w = (const float*)d_in[4];
  const float* k1b = (const float*)d_in[5];
  const float* v1w = (const float*)d_in[6];
  const float* v1b = (const float*)d_in[7];
  const float* r1w = (const float*)d_in[8];
  const float* r1b = (const float*)d_in[9];
  const float* q2w = (const float*)d_in[10];
  const float* q2b = (const float*)d_in[11];
  const float* k2w = (const float*)d_in[12];
  const float* k2b = (const float*)d_in[13];
  const float* v2w = (const float*)d_in[14];
  const float* v2b = (const float*)d_in[15];
  const float* r2w = (const float*)d_in[16];
  const float* r2b = (const float*)d_in[17];
  const float* cw  = (const float*)d_in[18];
  const float* cb  = (const float*)d_in[19];
  float* out = (float*)d_out;

  float* gK = (float*)d_ws;          // [16][64]
  float* gV = gK + 16 * 64;          // [16][64]
  float* gM = gV + 16 * 64;          // [16][4096]
  const int zn = 16 * 64 * 2 + 16 * 4096;

  zero_ws<<<(zn + 255) / 256, 256, 0, stream>>>(gK, zn);

  p1_kv_reduce<<<dim3(16 * WG_PER_TGT), dim3(256), 0, stream>>>(
      x1, x2, k1w, k1b, v1w, v1b, k2w, k2b, v2w, v2b, gK, gV, gM);

  (void)hipFuncSetAttribute(reinterpret_cast<const void*>(p2_fused),
                            hipFuncAttributeMaxDynamicSharedMemorySize, (int)SMEM_P2);
  p2_fused<<<dim3(16, 16, 8), dim3(256), SMEM_P2, stream>>>(
      x1, x2, q1w, q1b, r1w, r1b, q2w, q2b, r2w, r2b,
      gK, gV, gM, cw, cb, out);
}
